// BrainGNN_28278064677287
// MI455X (gfx1250) — matrix operand order repaired and checked by reading, not compiled, not run on hardware
//
#include <hip/hip_runtime.h>
#include <math.h>
#include <cstdint>

typedef __attribute__((ext_vector_type(16))) _Float16 v16h;
typedef __attribute__((ext_vector_type(8)))  _Float16 v8h;
typedef __attribute__((ext_vector_type(4)))  _Float16 v4h;
typedef __attribute__((ext_vector_type(16))) float    v16f;
typedef __attribute__((ext_vector_type(8)))  float    v8f;
typedef __attribute__((ext_vector_type(4)))  float    v4f;
typedef int v4i __attribute__((vector_size(16)));   // matches builtin proto

// CDNA5 async-to-LDS path (guarded: falls back to register staging if the
// toolchain doesn't declare these builtins).
#if __has_builtin(__builtin_amdgcn_global_load_async_to_lds_b128) && \
    __has_builtin(__builtin_amdgcn_s_wait_asynccnt)
#define USE_ASYNC_LDS 1
#else
#define USE_ASYNC_LDS 0
#endif

constexpr int GN    = 16384;    // nodes
constexpr int GH    = 32;       // hidden
constexpr int RPB   = 64;       // rows per block (4 waves x 16) -> grid = 256
constexpr int NTHR  = 128;      // 4 waves
constexpr int KC    = 32;       // K chunk per WMMA step
constexpr int NCH   = GN / KC;  // 512
constexpr int ASTRF = 36;       // f32 A-tile stride (144B: 16B-aligned, conflict-free)
constexpr int ASTRH = 40;       // f16 A-tile stride (80B)
constexpr int BSTR  = 40;       // f16 B-tile stride (transposed [col][k])
constexpr int MSTR  = 33;       // f32 m-tile stride

// ---------------------------------------------------------------------------
// Fold  x = m@Wm^T + bm ; gi = x@W_ih^T + b_ih  into  gi = m@W_eff^T + b_eff
// ---------------------------------------------------------------------------
__global__ void braingnn_weff_setup(const float* __restrict__ Wm,
                                    const float* __restrict__ bm,
                                    const float* __restrict__ Wih,
                                    const float* __restrict__ bih,
                                    float* __restrict__ weff,
                                    float* __restrict__ beff) {
    const int j = threadIdx.x;
    if (j >= 96) return;
    float be = bih[j];
    for (int t = 0; t < 32; ++t) be += Wih[j * 32 + t] * bm[t];
    beff[j] = be;
    for (int k = 0; k < 32; ++k) {
        float s = 0.0f;
        for (int t = 0; t < 32; ++t) s += Wih[j * 32 + t] * Wm[t * 32 + k];
        weff[j * 32 + k] = s;
    }
}

#if USE_ASYNC_LDS
__device__ __forceinline__ void async_cp_b128(const float* g, float* l) {
    __builtin_amdgcn_global_load_async_to_lds_b128(
        (__attribute__((address_space(1))) v4i*)(uintptr_t)g,
        (__attribute__((address_space(3))) v4i*)(uintptr_t)l,
        0, 0);
}
#endif

__device__ __forceinline__ v8f wmma_f16(v16h a, v16h b, v8f c) {
    return __builtin_amdgcn_wmma_f32_16x16x32_f16(false, a, false, b, (short)0, c, false, false);
}

// ---------------------------------------------------------------------------
// Fused: m = adj @ node_state (f16 WMMA, fp32 accum), then GRU epilogue.
// ---------------------------------------------------------------------------
__global__ __launch_bounds__(NTHR) void braingnn_fused(
    const float* __restrict__ adj,   // (GN, GN)
    const float* __restrict__ ns,    // (GN, GH)
    const float* __restrict__ Whh,   // (96, 32)
    const float* __restrict__ bhh,   // (96,)
    const float* __restrict__ weff,  // (96, 32)
    const float* __restrict__ beff,  // (96,)
    float* __restrict__ out)         // (GN, GH)
{
#if USE_ASYNC_LDS
    __shared__ float    ldsA[2][RPB * ASTRF];   // adj tile f32, filled by async DMA
#else
    __shared__ _Float16 ldsA[2][RPB * ASTRH];   // adj tile f16, reg-staged
#endif
    __shared__ _Float16 ldsB[2][GH * BSTR];     // node_state tile f16, TRANSPOSED [col][k]
    __shared__ float    ldsM[RPB * MSTR];       // m tile for epilogue

    const int tid  = threadIdx.x;
    const int lane = tid & 31;        // wave32
    const int wave = tid >> 5;        // 0..3
    const int rowBase = blockIdx.x * RPB;

    const int colq = (tid & 7) * 4;   // 0..28
    const int rowq = tid >> 3;        // 0..15

    // ---- B chunk staging (both paths): 32x32 f32 -> f16 transposed ----
    v4f rb0, rb1;
    auto loadB = [&](int k) {
        rb0 = *(const v4f*)(ns + (size_t)(k + rowq) * GH + colq);
        rb1 = *(const v4f*)(ns + (size_t)(k + 16 + rowq) * GH + colq);
    };
    auto storeB = [&](int p) {
#pragma unroll
        for (int j = 0; j < 4; ++j) {
            ldsB[p][(colq + j) * BSTR + rowq]      = (_Float16)rb0[j];
            ldsB[p][(colq + j) * BSTR + 16 + rowq] = (_Float16)rb1[j];
        }
    };

#if USE_ASYNC_LDS
    // ---- A chunk via async DMA: 64 rows x 32 k, f32, no VGPR staging ----
    auto issueA = [&](int p, int k) {
#pragma unroll
        for (int i = 0; i < 4; ++i) {
            const int row = i * 16 + rowq;
            async_cp_b128(adj + (size_t)(rowBase + row) * GN + k + colq,
                          &ldsA[p][row * ASTRF + colq]);
        }
    };
#else
    v4f ra[4];
    auto loadA = [&](int k) {
#pragma unroll
        for (int i = 0; i < 4; ++i)
            ra[i] = *(const v4f*)(adj + (size_t)(rowBase + i * 16 + rowq) * GN + k + colq);
    };
    auto storeA = [&](int p) {
#pragma unroll
        for (int i = 0; i < 4; ++i) {
            v4h h;
#pragma unroll
            for (int j = 0; j < 4; ++j) h[j] = (_Float16)ra[i][j];
            *(v4h*)(&ldsA[p][(size_t)(i * 16 + rowq) * ASTRH + colq]) = h;  // 8B aligned
        }
    };
#endif

    // Fragment addressing per CDNA5 wave32 WMMA VGPR layouts:
    // A 16x32 f16: lane&15 = M; lanes<16: K {0..7,16..23}, lanes>=16: K {8..15,24..31}
    // B 32x16 f16: lane&15 = N; lanes<16: K {0..7,16..23}, lanes>=16: K {8..15,24..31}
    const int frow = wave * 16 + (lane & 15);
    const int aseg = (lane >> 4) * 8;
    const int bcol = lane & 15;
    const int bko  = (lane >> 4) * 8;

    auto loadAfrag = [&](int p) -> v16h {
#if USE_ASYNC_LDS
        const float* base = &ldsA[p][frow * ASTRF];
        v4f a0 = *(const v4f*)(base + aseg);
        v4f a1 = *(const v4f*)(base + aseg + 4);
        v4f a2 = *(const v4f*)(base + 16 + aseg);
        v4f a3 = *(const v4f*)(base + 16 + aseg + 4);
        v8f  lo   = __builtin_shufflevector(a0, a1, 0,1,2,3,4,5,6,7);
        v8f  hi   = __builtin_shufflevector(a2, a3, 0,1,2,3,4,5,6,7);
        v16f full = __builtin_shufflevector(lo, hi, 0,1,2,3,4,5,6,7,8,9,10,11,12,13,14,15);
        return __builtin_convertvector(full, v16h);
#else
        v8h lo = *(const v8h*)(&ldsA[p][frow * ASTRH + aseg]);
        v8h hi = *(const v8h*)(&ldsA[p][frow * ASTRH + 16 + aseg]);
        return __builtin_shufflevector(lo, hi, 0,1,2,3,4,5,6,7,8,9,10,11,12,13,14,15);
#endif
    };
    auto loadBfrag = [&](int p, int cb) -> v16h {
        v8h lo = *(const v8h*)(&ldsB[p][(cb + bcol) * BSTR + bko]);
        v8h hi = *(const v8h*)(&ldsB[p][(cb + bcol) * BSTR + bko + 16]);
        return __builtin_shufflevector(lo, hi, 0,1,2,3,4,5,6,7,8,9,10,11,12,13,14,15);
    };

    v8f acc0 = {0.f,0.f,0.f,0.f,0.f,0.f,0.f,0.f};   // cols 0..15
    v8f acc1 = {0.f,0.f,0.f,0.f,0.f,0.f,0.f,0.f};   // cols 16..31

    // Single barrier per chunk: iter kc reads buf[kc&1]; writes target buf[(kc+1)&1],
    // whose readers (iter kc-1) drained their ds reads (s_wait_dscnt before the WMMA)
    // before signaling the previous barrier.
    loadB(0);
#if USE_ASYNC_LDS
    issueA(0, 0);
    for (int kc = 0; kc < NCH; ++kc) {
        const int p = kc & 1;
        storeB(p);
        __builtin_amdgcn_s_wait_asynccnt(0);     // this wave's buf-p DMAs complete
        __syncthreads();                         // all waves' stores/DMAs visible
        if (kc + 1 < NCH) { issueA(p ^ 1, (kc + 1) * KC); loadB((kc + 1) * KC); }
        v16h af  = loadAfrag(p);
        v16h bf0 = loadBfrag(p, 0);
        v16h bf1 = loadBfrag(p, 16);
        acc0 = wmma_f16(af, bf0, acc0);
        acc1 = wmma_f16(af, bf1, acc1);
    }
#else
    loadA(0);
    for (int kc = 0; kc < NCH; ++kc) {
        const int p = kc & 1;
        storeB(p);
        storeA(p);
        __syncthreads();
        if (kc + 1 < NCH) { loadA((kc + 1) * KC); loadB((kc + 1) * KC); }
        if (kc + 4 < NCH) {   // each 32-float row chunk is exactly one 128B line
            const int kpf = (kc + 4) * KC;
#pragma unroll
            for (int i = 0; i < 4; ++i)
                __builtin_prefetch(adj + (size_t)(rowBase + i * 16 + rowq) * GN + kpf + colq, 0, 3);
        }
        v16h af  = loadAfrag(p);
        v16h bf0 = loadBfrag(p, 0);
        v16h bf1 = loadBfrag(p, 16);
        acc0 = wmma_f16(af, bf0, acc0);
        acc1 = wmma_f16(af, bf1, acc1);
    }
#endif

    // ---- spill m tile (C layout: VGPR i -> M = i (lanes<16) / i+8 (lanes>=16)) ----
    {
        const int c    = lane & 15;
        const int rofs = (lane < 16) ? 0 : 8;
#pragma unroll
        for (int i = 0; i < 8; ++i) {
            const int rl = wave * 16 + rofs + i;
            ldsM[rl * MSTR + c]      = acc0[i];
            ldsM[rl * MSTR + 16 + c] = acc1[i];
        }
    }
    __syncthreads();

    // ---- GRU epilogue: 2 threads per row, 16 output cols each ----
    {
        const int row  = tid >> 1;        // 0..63
        const int half = tid & 1;
        const int grow = rowBase + row;

        float mrow[32], hrow[32];
#pragma unroll
        for (int t = 0; t < 32; ++t) mrow[t] = ldsM[row * MSTR + t];
#pragma unroll
        for (int t = 0; t < 32; t += 4) {
            v4f hv = *(const v4f*)(ns + (size_t)grow * GH + t);
            hrow[t] = hv[0]; hrow[t+1] = hv[1]; hrow[t+2] = hv[2]; hrow[t+3] = hv[3];
        }

        for (int cc = 0; cc < 16; ++cc) {
            const int col = half * 16 + cc;
            float gir = beff[col], giz = beff[32 + col], gin = beff[64 + col];
            float ghr = bhh[col],  ghz = bhh[32 + col],  ghn = bhh[64 + col];
#pragma unroll
            for (int t = 0; t < 32; ++t) {   // contiguous -> merged into b128 loads
                const float mv = mrow[t], hv = hrow[t];
                gir += mv * weff[col        * 32 + t];
                giz += mv * weff[(32 + col) * 32 + t];
                gin += mv * weff[(64 + col) * 32 + t];
                ghr += hv * Whh[col        * 32 + t];
                ghz += hv * Whh[(32 + col) * 32 + t];
                ghn += hv * Whh[(64 + col) * 32 + t];
            }
            const float r = 1.0f / (1.0f + __expf(-(gir + ghr)));
            const float z = 1.0f / (1.0f + __expf(-(giz + ghz)));
            const float n = tanhf(gin + r * ghn);
            out[(size_t)grow * GH + col] = (1.0f - z) * n + z * hrow[col];
        }
    }
}

// ---------------------------------------------------------------------------
extern "C" void kernel_launch(void* const* d_in, const int* in_sizes, int n_in,
                              void* d_out, int out_size, void* d_ws, size_t ws_size,
                              hipStream_t stream) {
    (void)in_sizes; (void)n_in; (void)out_size; (void)ws_size;
    const float* adj = (const float*)d_in[0];
    const float* ns  = (const float*)d_in[1];
    const float* Wm  = (const float*)d_in[2];
    const float* bm  = (const float*)d_in[3];
    const float* Wih = (const float*)d_in[4];
    const float* Whh = (const float*)d_in[5];
    const float* bih = (const float*)d_in[6];
    const float* bhh = (const float*)d_in[7];
    float* out  = (float*)d_out;
    float* weff = (float*)d_ws;          // 96*32 floats
    float* beff = weff + 96 * 32;        // 96 floats

    braingnn_weff_setup<<<1, 128, 0, stream>>>(Wm, bm, Wih, bih, weff, beff);
    braingnn_fused<<<GN / RPB, NTHR, 0, stream>>>(adj, ns, Whh, bhh, weff, beff, out);
}
